// StaticFusionEncoder_764504179158
// MI455X (gfx1250) — compile-verified, matrix operand order checked by reading, not run on hardware
//
#include <hip/hip_runtime.h>
#include <hip/hip_bf16.h>
#include <math.h>

typedef _Float16 v16h __attribute__((ext_vector_type(16)));
typedef _Float16 v8h  __attribute__((ext_vector_type(8)));
typedef float    v8f  __attribute__((ext_vector_type(8)));

#define DIM     256
#define HIDDEN  192
#define BLK_M   32      // rows per block
#define NWAVES  12      // one wave per 16-col N tile (12*16 = 192)
#define NTHREADS (NWAVES * 32)

// ---------------------------------------------------------------------------
// Kernel 0: convert + transpose weights to f16 in workspace.
//   w1t[h][d] = (f16) w1[d][h]   (192 x 256)
//   w2t[k][h] = (f16) w2[h][k]   (192 x 192)
// ---------------------------------------------------------------------------
__global__ void convert_weights(const float* __restrict__ w1,
                                const float* __restrict__ w2,
                                _Float16* __restrict__ w1t,
                                _Float16* __restrict__ w2t) {
    int i = blockIdx.x * blockDim.x + threadIdx.x;
    const int n1 = HIDDEN * DIM;      // 49152
    const int n2 = HIDDEN * HIDDEN;   // 36864
    if (i < n1) {
        int h = i / DIM, d = i - h * DIM;
        w1t[i] = (_Float16)w1[d * HIDDEN + h];
    } else if (i < n1 + n2) {
        int j = i - n1;
        int k = j / HIDDEN, h = j - k * HIDDEN;
        w2t[j] = (_Float16)w2[h * HIDDEN + k];
    }
}

// Load a 16x32 (f16) A-style fragment for one lane from a row-major array.
// row  : pointer to start of this lane's row (M for A, N for B^T)
// kb   : K base for this lane = kc*32 + ((lane&16) ? 8 : 0)
__device__ __forceinline__ v16h load_frag16(const _Float16* row, int kb) {
    v8h lo = *(const v8h*)(row + kb);
    v8h hi = *(const v8h*)(row + kb + 16);
    v16h f;
#pragma unroll
    for (int i = 0; i < 8; ++i) { f[i] = lo[i]; f[i + 8] = hi[i]; }
    return f;
}

__device__ __forceinline__ float gelu_exact(float v) {
    return 0.5f * v * (1.0f + erff(v * 0.70710678118654752f));
}

// ---------------------------------------------------------------------------
// Kernel 1: fused MLP + mask + pos.
// grid.x = rows / BLK_M, block = 384 threads (12 waves).
// ---------------------------------------------------------------------------
__global__ __launch_bounds__(NTHREADS)
void mlp_fused(const float* __restrict__ x,
               const float* __restrict__ b1,
               const float* __restrict__ b2,
               const _Float16* __restrict__ w1t,
               const _Float16* __restrict__ w2t,
               float* __restrict__ yout,
               float* __restrict__ maskout,
               float* __restrict__ posout) {
    __shared__ __align__(16) _Float16 xs[BLK_M * DIM];     // 16 KB
    __shared__ __align__(16) _Float16 hs[BLK_M * HIDDEN];  // 12 KB
    __shared__ float validf[BLK_M];

    const int tid  = threadIdx.x;
    const int r0   = blockIdx.x * BLK_M;
    const int lane = tid & 31;
    const int wave = tid >> 5;           // == N tile index, 0..11
    const int ln   = lane & 15;          // lane's column/row within tile
    const int ksel = (lane & 16) ? 8 : 0;
    const int mo   = (lane & 16) ? 8 : 0;

    // ---- stage x tile (f32 -> f16) into LDS --------------------------------
    {
        const float4* xg = (const float4*)(x + (size_t)r0 * DIM);
        for (int i = tid; i < BLK_M * (DIM / 4); i += NTHREADS) {
            float4 v = xg[i];
            int row = i >> 6;            // DIM/4 = 64 float4 per row
            int c4  = (i & 63) << 2;
            _Float16* p = &xs[row * DIM + c4];
            p[0] = (_Float16)v.x; p[1] = (_Float16)v.y;
            p[2] = (_Float16)v.z; p[3] = (_Float16)v.w;
        }
    }

    // ---- mask_p, valid flag, pos (one thread per row, exact f32 check) -----
    if (tid < BLK_M) {
        size_t g = (size_t)(r0 + tid);
        const float* xr = x + g * DIM;
        int nz = 0;
#pragma unroll
        for (int j = 0; j < 10; ++j) nz += (xr[j] != 0.0f);
        bool maskp = (nz == 0);
        validf[tid] = maskp ? 0.0f : 1.0f;
        maskout[g]  = maskp ? 1.0f : 0.0f;
        float* pp = posout + g * 7;
        pp[0] = xr[0]; pp[1] = xr[1]; pp[2] = xr[2]; pp[3] = xr[3];
        pp[4] = 0.0f;  pp[5] = 1.0f;  pp[6] = 0.0f;
    }
    __syncthreads();

    // ---- GEMM1: [32 x 256] x [256 x 16]  (this wave's N tile) --------------
    const _Float16* w1row = w1t + (size_t)(wave * 16 + ln) * DIM;  // B = row of W1^T
    v8f c0 = {}, c1 = {};
#pragma unroll
    for (int kc = 0; kc < DIM / 32; ++kc) {
        int kb = kc * 32 + ksel;
        v16h b  = load_frag16(w1row, kb);
        v16h a0 = load_frag16(&xs[ln * DIM], kb);
        v16h a1 = load_frag16(&xs[(16 + ln) * DIM], kb);
        c0 = __builtin_amdgcn_wmma_f32_16x16x32_f16(false, a0, false, b,
                                                    (short)0, c0, false, false);
        c1 = __builtin_amdgcn_wmma_f32_16x16x32_f16(false, a1, false, b,
                                                    (short)0, c1, false, false);
    }

    // ---- bias + exact gelu, h -> LDS (f16) ---------------------------------
    {
        const int n = wave * 16 + ln;
        const float bias1 = b1[n];
#pragma unroll
        for (int r = 0; r < 8; ++r) {
            hs[(r + mo) * HIDDEN + n]      = (_Float16)gelu_exact(c0[r] + bias1);
            hs[(16 + r + mo) * HIDDEN + n] = (_Float16)gelu_exact(c1[r] + bias1);
        }
    }
    __syncthreads();

    // ---- GEMM2: [32 x 192] x [192 x 16] ------------------------------------
    const _Float16* w2row = w2t + (size_t)(wave * 16 + ln) * HIDDEN;
    v8f d0 = {}, d1 = {};
#pragma unroll
    for (int kc = 0; kc < HIDDEN / 32; ++kc) {
        int kb = kc * 32 + ksel;
        v16h b  = load_frag16(w2row, kb);
        v16h a0 = load_frag16(&hs[ln * HIDDEN], kb);
        v16h a1 = load_frag16(&hs[(16 + ln) * HIDDEN], kb);
        d0 = __builtin_amdgcn_wmma_f32_16x16x32_f16(false, a0, false, b,
                                                    (short)0, d0, false, false);
        d1 = __builtin_amdgcn_wmma_f32_16x16x32_f16(false, a1, false, b,
                                                    (short)0, d1, false, false);
    }

    // ---- bias + mask select + store ---------------------------------------
    {
        const int n = wave * 16 + ln;
        const float bias2 = b2[n];
#pragma unroll
        for (int r = 0; r < 8; ++r) {
            int m = r + mo;
            float v = (validf[m] != 0.0f) ? (d0[r] + bias2) : 0.0f;
            yout[(size_t)(r0 + m) * HIDDEN + n] = v;
            m = 16 + r + mo;
            v = (validf[m] != 0.0f) ? (d1[r] + bias2) : 0.0f;
            yout[(size_t)(r0 + m) * HIDDEN + n] = v;
        }
    }
}

// ---------------------------------------------------------------------------
extern "C" void kernel_launch(void* const* d_in, const int* in_sizes, int n_in,
                              void* d_out, int out_size, void* d_ws, size_t ws_size,
                              hipStream_t stream) {
    const float* x  = (const float*)d_in[0];
    const float* w1 = (const float*)d_in[1];
    const float* b1 = (const float*)d_in[2];
    const float* w2 = (const float*)d_in[3];
    const float* b2 = (const float*)d_in[4];

    const int rows = in_sizes[0] / DIM;   // B*P = 65536

    _Float16* w1t = (_Float16*)d_ws;                    // 192*256 halves
    _Float16* w2t = w1t + (size_t)HIDDEN * DIM;         // 192*192 halves

    float* yout    = (float*)d_out;                     // rows * 192
    float* maskout = yout + (size_t)rows * HIDDEN;      // rows
    float* posout  = maskout + rows;                    // rows * 7

    {
        const int total = HIDDEN * DIM + HIDDEN * HIDDEN;
        convert_weights<<<(total + 255) / 256, 256, 0, stream>>>(w1, w2, w1t, w2t);
    }

    mlp_fused<<<rows / BLK_M, NTHREADS, 0, stream>>>(x, b1, b2, w1t, w2t,
                                                     yout, maskout, posout);
}
